// Pointnet2_16776142258207
// MI455X (gfx1250) — matrix operand order repaired and checked by reading, not compiled, not run on hardware
//
#include <hip/hip_runtime.h>

// ---------------------------------------------------------------------------
// Types for CDNA5 WMMA (wave32): D(f32 16x16) = A(bf16 16x32) * B(bf16 32x16) + C
// ---------------------------------------------------------------------------
typedef __attribute__((ext_vector_type(16))) __bf16 v16bf;
typedef __attribute__((ext_vector_type(8)))  float  v8f;

__device__ __forceinline__ __bf16 f2bf(float f) {
  unsigned u; __builtin_memcpy(&u, &f, 4);
  unsigned r = u + 0x7fffu + ((u >> 16) & 1u);   // round-to-nearest-even
  unsigned short h = (unsigned short)(r >> 16);
  __bf16 o; __builtin_memcpy(&o, &h, 2);
  return o;
}
__device__ __forceinline__ float bf2f(__bf16 h) {
  unsigned short s; __builtin_memcpy(&s, &h, 2);
  unsigned u = ((unsigned)s) << 16;
  float f; __builtin_memcpy(&f, &u, 4);
  return f;
}

struct Layer {
  const __bf16* Wt;   // converted weights, K-major: [cout][cinPad] bf16, zero padded
  const float*  b;
  const float*  g;
  const float*  be;
  int cinPad;         // cin rounded up to multiple of 32
  int cout;           // multiple of 32
};
struct Mlp { Layer l[3]; int n; };

// A fragment: 16x32 bf16 tile from LDS activations (row-major, stride lda).
// Layout: lanes 0-15 -> M=lane, K in {0..15}; lanes 16-31 -> K in {16..31} (matches load_B).
__device__ __forceinline__ v16bf load_A(const __bf16* act, int lda, int mbase, int kbase) {
  int lane = threadIdx.x & 31;
  const __bf16* row = act + (mbase + (lane & 15)) * lda;
  int h8 = (lane >> 4) << 4;  // 0 or 16
  v16bf a;
#pragma unroll
  for (int j = 0; j < 4; ++j) {
    int k0 = kbase + h8 + 2 * j;
    a[2*j]     = row[k0];
    a[2*j+1]   = row[k0 + 1];
    int k1 = kbase + 8 + h8 + 2 * j;
    a[8+2*j]   = row[k1];
    a[8+2*j+1] = row[k1 + 1];
  }
  return a;
}

// B fragment: 32x16 bf16 tile; lane L holds N=L%16, K = kbase + 16*(L/16) .. +15.
// Wt is K-major so this is one contiguous 32-byte load per lane.
__device__ __forceinline__ v16bf load_B(const __bf16* Wt, int cinPad, int nbase, int kbase) {
  int lane = threadIdx.x & 31;
  const __bf16* p = Wt + (size_t)(nbase + (lane & 15)) * cinPad + kbase + ((lane >> 4) << 4);
  v16bf r; __builtin_memcpy(&r, p, 32);
  return r;
}

// One conv1x1+BN(affine)+ReLU layer over 32 rows of activations, via WMMA.
// Tile loop is wave-uniform -> EXEC all-ones at every WMMA.
__device__ __forceinline__ void mlp_layer(const __bf16* in, int lda, __bf16* out, int ldo, Layer L) {
  int wave = threadIdx.x >> 5;
  int nw   = blockDim.x >> 5;
  int lane = threadIdx.x & 31;
  int ntiles = 2 * (L.cout >> 4);            // 2 M-tiles (32 rows) x cout/16 N-tiles
  for (int t = wave; t < ntiles; t += nw) {
    int mbase = (t & 1) << 4;
    int nbase = (t >> 1) << 4;
    v8f acc = {0.f, 0.f, 0.f, 0.f, 0.f, 0.f, 0.f, 0.f};
    for (int k = 0; k < L.cinPad; k += 32) {
      v16bf a  = load_A(in, lda, mbase, k);
      v16bf bm = load_B(L.Wt, L.cinPad, nbase, k);
      acc = __builtin_amdgcn_wmma_f32_16x16x32_bf16(false, a, false, bm,
                                                    (short)0, acc, false, false);
    }
    int n = nbase + (lane & 15);
    float bb = L.b[n], gg = L.g[n], ee = L.be[n];
    int mof = mbase + ((lane >> 4) << 3);
#pragma unroll
    for (int j = 0; j < 8; ++j) {
      float v = (acc[j] + bb) * gg + ee;
      v = v > 0.f ? v : 0.f;
      out[(mof + j) * ldo + n] = f2bf(v);
    }
  }
}

// Run full MLP stack, ping-ponging between buf0 (holds input) and buf1.
__device__ __forceinline__ const __bf16* run_mlp(__bf16* buf0, int lda0, __bf16* buf1, Mlp M) {
  const __bf16* in = buf0; int lda = lda0;
  const __bf16* last = buf0;
  for (int i = 0; i < M.n; ++i) {
    __syncthreads();
    __bf16* out = (i & 1) ? buf0 : buf1;
    mlp_layer(in, lda, out, M.l[i].cout, M.l[i]);
    in = out; lda = M.l[i].cout; last = out;
  }
  __syncthreads();
  return last;
}

// ---------------------------------------------------------------------------
// Weight conversion: f32 [cin][cout] -> bf16 K-major [cout][cinPad], zero-pad K.
// ---------------------------------------------------------------------------
__global__ void conv_w_kernel(const float* __restrict__ W, __bf16* __restrict__ Wt,
                              int cin, int cout, int cinPad) {
  int t = blockIdx.x * blockDim.x + threadIdx.x;
  int total = cout * cinPad;
  if (t >= total) return;
  int n = t / cinPad, k = t - n * cinPad;
  Wt[t] = f2bf(k < cin ? W[(size_t)k * cout + n] : 0.f);
}

// ---------------------------------------------------------------------------
// Farthest point sampling: one block per batch, sequential scan + LDS argmax.
// ---------------------------------------------------------------------------
__global__ void fps_kernel(const float* __restrict__ xyz, int* __restrict__ outIdx,
                           int N, int S) {
  __shared__ float dists[4096];
  __shared__ float redv[256];
  __shared__ int   redi[256];
  __shared__ int   lastIdx;
  int b = blockIdx.x;
  const float* base = xyz + (size_t)b * N * 3;
  int tid = threadIdx.x, nt = blockDim.x;
  for (int i = tid; i < N; i += nt) dists[i] = 1e10f;
  if (tid == 0) { lastIdx = 0; outIdx[(size_t)b * S] = 0; }
  __syncthreads();
  for (int it = 1; it < S; ++it) {
    int l = lastIdx;
    float lx = base[l*3], ly = base[l*3+1], lz = base[l*3+2];
    float bestv = -1.f; int besti = 0;
    for (int i = tid; i < N; i += nt) {
      float dx = base[i*3] - lx, dy = base[i*3+1] - ly, dz = base[i*3+2] - lz;
      float d = dx*dx + dy*dy + dz*dz;
      float m = fminf(dists[i], d);
      dists[i] = m;
      if (m > bestv) { bestv = m; besti = i; }
    }
    redv[tid] = bestv; redi[tid] = besti;
    __syncthreads();
    for (int off = nt >> 1; off > 0; off >>= 1) {
      if (tid < off && redv[tid + off] > redv[tid]) {
        redv[tid] = redv[tid + off]; redi[tid] = redi[tid + off];
      }
      __syncthreads();
    }
    if (tid == 0) { lastIdx = redi[0]; outIdx[(size_t)b * S + it] = redi[0]; }
    __syncthreads();
  }
}

__global__ void gather_xyz_kernel(const float* __restrict__ xyz, const int* __restrict__ idx,
                                  float* __restrict__ nx, int N, int S, int T) {
  int t = blockIdx.x * blockDim.x + threadIdx.x;
  if (t >= T) return;
  int b = t / S;
  int p = idx[t];
  const float* q = xyz + ((size_t)b * N + p) * 3;
  nx[t*3+0] = q[0]; nx[t*3+1] = q[1]; nx[t*3+2] = q[2];
}

// ---------------------------------------------------------------------------
// Ball query: one wave32 per query; first 32 in-radius hits in index order,
// padded with the first hit (always exists: center itself).
// ---------------------------------------------------------------------------
__global__ void ball_kernel(const float* __restrict__ xyz, const float* __restrict__ nx,
                            int* __restrict__ ball, int N, int S, float r2, int T) {
  int wid = blockIdx.x * (blockDim.x >> 5) + (threadIdx.x >> 5);
  if (wid >= T) return;
  int lane = threadIdx.x & 31;
  int b = wid / S;
  const float* base = xyz + (size_t)b * N * 3;
  float cx = nx[wid*3], cy = nx[wid*3+1], cz = nx[wid*3+2];
  int* out = ball + (size_t)wid * 32;
  int count = 0, first = -1;
  for (int st = 0; st < N; st += 32) {
    int p = st + lane;
    float dx = base[p*3] - cx, dy = base[p*3+1] - cy, dz = base[p*3+2] - cz;
    bool ok = (dx*dx + dy*dy + dz*dz) < r2;
    unsigned mask = (unsigned)__ballot(ok);
    if (first < 0 && mask) first = st + __ffs(mask) - 1;
    int pos = count + __popc(mask & ((1u << lane) - 1u));
    if (ok && pos < 32) out[pos] = p;
    count += __popc(mask);
    if (count >= 32) break;
  }
  if (first < 0) first = 0;
  if (count < 32)
    for (int i = count + lane; i < 32; i += 32) out[i] = first;
}

// ---------------------------------------------------------------------------
// SA: gather group (rel-xyz ++ feats) into LDS bf16, run MLP via WMMA, maxpool K=32.
// One block per (batch, sampled point). dyn LDS: 32*cinPad + 32*maxcout bf16.
// ---------------------------------------------------------------------------
__global__ void sa_kernel(const float* __restrict__ xyz, const float* __restrict__ feats, int Cf,
                          const float* __restrict__ nx, const int* __restrict__ ball,
                          float* __restrict__ outF, int N, int S, int cin, int cinPad, Mlp M) {
  extern __shared__ __bf16 smem_dyn[];
  __bf16* buf0 = smem_dyn;
  __bf16* buf1 = smem_dyn + 32 * cinPad;
  int bs = blockIdx.x;
  int b = bs / S;
  const int* bi = ball + (size_t)bs * 32;
  float cx = nx[bs*3], cy = nx[bs*3+1], cz = nx[bs*3+2];
  for (int t = threadIdx.x; t < 32 * cinPad; t += blockDim.x) {
    int k = t / cinPad, c = t - k * cinPad;
    int p = bi[k];
    float v = 0.f;
    if (c < 3) {
      float q = xyz[((size_t)b * N + p) * 3 + c];
      v = q - (c == 0 ? cx : (c == 1 ? cy : cz));
    } else if (c < cin) {
      v = feats[((size_t)b * N + p) * Cf + (c - 3)];
    }
    buf0[t] = f2bf(v);
  }
  const __bf16* last = run_mlp(buf0, cinPad, buf1, M);
  int cl = M.l[M.n - 1].cout;
  for (int c = threadIdx.x; c < cl; c += blockDim.x) {
    float m = -3e38f;
    for (int k = 0; k < 32; ++k) {
      float v = bf2f(last[k * cl + c]);
      m = v > m ? v : m;
    }
    outF[(size_t)bs * cl + c] = m;
  }
}

// ---------------------------------------------------------------------------
// FP: 3-NN inverse-distance interpolation + concat + MLP via WMMA.
// One block per 32 dense points. NN scratch overlays buf1 (free during fill).
// ---------------------------------------------------------------------------
__global__ void fp_kernel(const float* __restrict__ xyz1, const float* __restrict__ xyz2,
                          const float* __restrict__ f1, int C1,
                          const float* __restrict__ f2, int C2,
                          float* __restrict__ outF, int N1, int N2, int cinPad, Mlp M) {
  extern __shared__ __bf16 smem_dyn[];
  __bf16* buf0 = smem_dyn;
  __bf16* buf1 = smem_dyn + 32 * cinPad;
  float* w3 = (float*)buf1;          // 32*3 floats, overlays buf1 during fill
  int*   i3 = (int*)(w3 + 96);       // 32*3 ints
  int tiles = N1 / 32;
  int b = blockIdx.x / tiles;
  int tile = blockIdx.x - b * tiles;
  int base1 = b * N1 + tile * 32;
  if (threadIdx.x < 32) {
    int p = base1 + threadIdx.x;
    float px = xyz1[p*3], py = xyz1[p*3+1], pz = xyz1[p*3+2];
    float d0 = 3e38f, d1 = 3e38f, d2 = 3e38f; int j0 = 0, j1 = 0, j2 = 0;
    for (int j = 0; j < N2; ++j) {
      const float* q = xyz2 + ((size_t)b * N2 + j) * 3;
      float dx = px - q[0], dy = py - q[1], dz = pz - q[2];
      float d = dx*dx + dy*dy + dz*dz;
      if (d < d0)      { d2 = d1; j2 = j1; d1 = d0; j1 = j0; d0 = d; j0 = j; }
      else if (d < d1) { d2 = d1; j2 = j1; d1 = d;  j1 = j; }
      else if (d < d2) { d2 = d;  j2 = j; }
    }
    float w0 = 1.f / (d0 + 1e-8f), w1 = 1.f / (d1 + 1e-8f), w2 = 1.f / (d2 + 1e-8f);
    float ws = w0 + w1 + w2;
    w3[threadIdx.x*3+0] = w0 / ws; w3[threadIdx.x*3+1] = w1 / ws; w3[threadIdx.x*3+2] = w2 / ws;
    i3[threadIdx.x*3+0] = j0;      i3[threadIdx.x*3+1] = j1;      i3[threadIdx.x*3+2] = j2;
  }
  __syncthreads();
  int cin = C1 + C2;
  for (int t = threadIdx.x; t < 32 * cinPad; t += blockDim.x) {
    int k = t / cinPad, c = t - k * cinPad;
    float v = 0.f;
    if (c < C2) {
      const float* fb = f2 + (size_t)b * N2 * C2 + c;
      v = w3[k*3+0] * fb[(size_t)i3[k*3+0] * C2]
        + w3[k*3+1] * fb[(size_t)i3[k*3+1] * C2]
        + w3[k*3+2] * fb[(size_t)i3[k*3+2] * C2];
    } else if (c < cin) {
      v = f1[(size_t)(base1 + k) * C1 + (c - C2)];
    }
    buf0[t] = f2bf(v);
  }
  const __bf16* last = run_mlp(buf0, cinPad, buf1, M);
  int cl = M.l[M.n - 1].cout;
  for (int t = threadIdx.x; t < 32 * cl; t += blockDim.x) {
    int k = t / cl, c = t - k * cl;
    outF[(size_t)(base1 + k) * cl + c] = bf2f(last[k * cl + c]);
  }
}

// ---------------------------------------------------------------------------
// Head: FC(128->128, affine+relu) via WMMA, then 13-class linear head (scalar GEMV).
// ---------------------------------------------------------------------------
__global__ void head_kernel(const float* __restrict__ f, Layer fc,
                            const float* __restrict__ Wout, const float* __restrict__ bout,
                            float* __restrict__ out, int N) {
  extern __shared__ __bf16 smem_dyn[];
  __bf16* buf0 = smem_dyn;
  __bf16* buf1 = smem_dyn + 32 * 128;
  int tiles = N / 32;
  int b = blockIdx.x / tiles;
  int tile = blockIdx.x - b * tiles;
  int base = b * N + tile * 32;
  for (int t = threadIdx.x; t < 32 * 128; t += blockDim.x)
    buf0[t] = f2bf(f[(size_t)base * 128 + t]);
  __syncthreads();
  mlp_layer(buf0, 128, buf1, 128, fc);
  __syncthreads();
  for (int t = threadIdx.x; t < 32 * 13; t += blockDim.x) {
    int k = t / 13, c = t - k * 13;
    float acc = bout[c];
    for (int j = 0; j < 128; ++j)
      acc += bf2f(buf1[k * 128 + j]) * Wout[(size_t)j * 13 + c];
    out[(size_t)(base + k) * 13 + c] = acc;
  }
}

// ---------------------------------------------------------------------------
// Host: workspace layout, weight conversion, pipeline launch (all on `stream`).
// ---------------------------------------------------------------------------
static const int   SA_DIMS[4][4] = {{9,32,32,64},{67,64,64,128},{131,128,128,256},{259,256,256,512}};
static const int   SA_S[4]       = {1024,256,64,16};
static const float SA_R[4]       = {0.1f,0.2f,0.4f,0.8f};
static const int   FP_DIMS_[4][4]= {{768,256,256,0},{384,256,256,0},{320,256,128,0},{134,128,128,128}};
static const int   FP_NL[4]      = {2,2,2,3};

extern "C" void kernel_launch(void* const* d_in, const int* in_sizes, int n_in,
                              void* d_out, int out_size, void* d_ws, size_t ws_size,
                              hipStream_t stream) {
  (void)in_sizes; (void)n_in; (void)out_size; (void)ws_size;
  const int B = 16, N0 = 4096;
  const float* xyz    = (const float*)d_in[0];
  const float* points = (const float*)d_in[1];

  char* wsb = (char*)d_ws;
  size_t off = 0;
  auto alloc = [&](size_t bytes) -> void* {
    off = (off + 255) & ~(size_t)255;
    void* p = wsb + off;
    off += bytes;
    return p;
  };

  // ---- convert all MLP weights to bf16 K-major, build Mlp descriptors ----
  int pi = 2;
  auto mkLayer = [&](int cin, int cout) -> Layer {
    int cp = (cin + 31) & ~31;
    __bf16* Wt = (__bf16*)alloc((size_t)cp * cout * sizeof(__bf16));
    int tot = cp * cout;
    conv_w_kernel<<<(tot + 255) / 256, 256, 0, stream>>>((const float*)d_in[pi], Wt, cin, cout, cp);
    Layer L;
    L.Wt = Wt;
    L.b  = (const float*)d_in[pi + 1];
    L.g  = (const float*)d_in[pi + 2];
    L.be = (const float*)d_in[pi + 3];
    L.cinPad = cp; L.cout = cout;
    pi += 4;
    return L;
  };

  Mlp saM[4];
  for (int l = 0; l < 4; ++l) {
    saM[l].n = 3;
    for (int i = 0; i < 3; ++i) saM[l].l[i] = mkLayer(SA_DIMS[l][i], SA_DIMS[l][i + 1]);
  }
  Mlp fpM[4];
  for (int l = 0; l < 4; ++l) {
    fpM[l].n = FP_NL[l];
    for (int i = 0; i < FP_NL[l]; ++i) fpM[l].l[i] = mkLayer(FP_DIMS_[l][i], FP_DIMS_[l][i + 1]);
    for (int i = FP_NL[l]; i < 3; ++i) fpM[l].l[i] = fpM[l].l[0];  // unused
  }
  Layer fcL = mkLayer(128, 128);
  const float* outW = (const float*)d_in[pi];
  const float* outB = (const float*)d_in[pi + 1];

  // ---- Set Abstraction levels ----
  const float* lx[5]; const float* lf[5]; int lN[5], lC[5];
  lx[0] = xyz; lf[0] = points; lN[0] = N0; lC[0] = 6;
  for (int l = 0; l < 4; ++l) {
    int S = SA_S[l], Np = lN[l];
    int T = B * S;
    int* fpsIdx = (int*)alloc((size_t)T * 4);
    float* nx   = (float*)alloc((size_t)T * 3 * 4);
    int* ball   = (int*)alloc((size_t)T * 32 * 4);
    int cout    = SA_DIMS[l][3];
    float* outF = (float*)alloc((size_t)T * cout * 4);

    fps_kernel<<<B, 256, 0, stream>>>(lx[l], fpsIdx, Np, S);
    gather_xyz_kernel<<<(T + 255) / 256, 256, 0, stream>>>(lx[l], fpsIdx, nx, Np, S, T);
    ball_kernel<<<(T + 3) / 4, 128, 0, stream>>>(lx[l], nx, ball, Np, S, SA_R[l] * SA_R[l], T);

    int cin = SA_DIMS[l][0];
    int cp = (cin + 31) & ~31;
    int maxc = 0;
    for (int i = 1; i < 4; ++i) if (SA_DIMS[l][i] > maxc) maxc = SA_DIMS[l][i];
    size_t sh = (size_t)32 * (cp + maxc) * sizeof(__bf16);
    sa_kernel<<<T, 128, sh, stream>>>(lx[l], lf[l], lC[l], nx, ball, outF, Np, S, cin, cp, saM[l]);

    lx[l + 1] = nx; lf[l + 1] = outF; lN[l + 1] = S; lC[l + 1] = cout;
  }

  // ---- Feature Propagation levels ----
  const float* f = lf[4]; int Cs = lC[4];
  for (int i = 0; i < 4; ++i) {
    int lvl = 3 - i;
    int N1 = lN[lvl], N2 = lN[lvl + 1], C1 = lC[lvl];
    int cin = FP_DIMS_[i][0];                    // == Cs + C1
    int cp = (cin + 31) & ~31;
    int nl = FP_NL[i];
    int cl = FP_DIMS_[i][nl];
    int maxc = 0;
    for (int j = 1; j <= nl; ++j) if (FP_DIMS_[i][j] > maxc) maxc = FP_DIMS_[i][j];
    float* outF = (float*)alloc((size_t)B * N1 * cl * 4);
    size_t sh = (size_t)32 * (cp + maxc) * sizeof(__bf16);
    fp_kernel<<<B * (N1 / 32), 128, sh, stream>>>(lx[lvl], lx[lvl + 1], lf[lvl], C1,
                                                  f, Cs, outF, N1, N2, cp, fpM[i]);
    f = outF; Cs = cl;
  }

  // ---- FC + classifier head -> d_out [16,4096,13] f32 ----
  size_t shH = (size_t)2 * 32 * 128 * sizeof(__bf16);
  head_kernel<<<B * (N0 / 32), 128, shH, stream>>>(f, fcL, outW, outB, (float*)d_out, N0);
}
